// Speech_Sampling_63075889709251
// MI455X (gfx1250) — matrix operand order repaired and checked by reading, not compiled, not run on hardware
//
#include <hip/hip_runtime.h>
#include <hip/hip_bf16.h>
#include <math.h>

typedef __bf16 bf16_t;
typedef bf16_t bf16x16 __attribute__((ext_vector_type(16)));
typedef bf16_t bf16x8  __attribute__((ext_vector_type(8)));
typedef float  v8f     __attribute__((ext_vector_type(8)));

// ---- WMMA helpers (CDNA5 wave32, v_wmma_f32_16x16x32_bf16) ----------------

__device__ __forceinline__ v8f wmma_bf16(bf16x16 a, bf16x16 b, v8f c) {
  // 8 args: (neg_a, A, neg_b, B, c_mod, C, reuse_a, reuse_b)
  return __builtin_amdgcn_wmma_f32_16x16x32_bf16(
      false, a, false, b, (short)0, c, false, false);
}

// Build a 16-element fragment from two contiguous 8-element (16B) chunks.
// A (16x32 bf16): lane holds K = {16*half+0..15}                     -> p, p+8
// B (32x16 bf16): lane holds K = {16*half+0..15}                     -> p, p+8
__device__ __forceinline__ bf16x16 frag2(const bf16_t* p0, const bf16_t* p1) {
  bf16x8 lo = *(const bf16x8*)p0;
  bf16x8 hi = *(const bf16x8*)p1;
  bf16x16 r;
#pragma unroll
  for (int i = 0; i < 8; ++i) { r[i] = lo[i]; r[i + 8] = hi[i]; }
  return r;
}

__device__ __forceinline__ float gelu_exact(float x) {
  return 0.5f * x * (1.0f + erff(x * 0.70710678118654752f));
}

// LDS row strides padded to avoid bank-0 pileups while keeping 16B alignment
#define XS_LD  136  // 18 x (128+8)
#define AS_LD  40   // 16 x (32+8)
#define BS_LD  40   // 128 x (32+8), transposed B tile [n][k]
#define HB_LD  520  // 16 x (512+8)

// ---- Stage 1: conv1 (K=3, 128->4096, stride 1, SAME) + GELU ---------------
// GEMM: M=16000 (b*2000+t), K=384 (k*128+cin), N=4096. Output bf16.
__global__ __launch_bounds__(256) void conv1_gelu_k(
    const float* __restrict__ x,   // [8,2000,128]
    const float* __restrict__ w1,  // [384,4096]
    const float* __restrict__ b1,  // [4096]
    bf16_t* __restrict__ h1)       // [8,2000,4096] bf16
{
  __shared__ __align__(16) bf16_t xs[18 * XS_LD];   // rows t0-1 .. t0+16
  __shared__ __align__(16) bf16_t bsh[128 * BS_LD]; // B tile transposed [n][k]
  const int lane = threadIdx.x & 31;
  const int wave = threadIdx.x >> 5;
  const int m0 = blockIdx.y * 16;
  const int b  = m0 / 2000;
  const int t0 = m0 % 2000;         // 2000 % 16 == 0: tiles never cross batch
  const int n0 = blockIdx.x * 128;
  const int row = lane & 15, half = lane >> 4;

  for (int idx = threadIdx.x; idx < 18 * 128; idx += 256) {
    int r = idx >> 7, cin = idx & 127;
    int tt = t0 - 1 + r;
    float v = (tt >= 0 && tt < 2000) ? x[((size_t)b * 2000 + tt) * 128 + cin] : 0.f;
    xs[r * XS_LD + cin] = (bf16_t)v;
  }

  v8f acc = {};
  for (int kc0 = 0; kc0 < 384; kc0 += 32) {
    __syncthreads();
    for (int idx = threadIdx.x; idx < 32 * 128; idx += 256) {
      int nn = idx & 127, kk = idx >> 7;
      bsh[nn * BS_LD + kk] = (bf16_t)w1[(size_t)(kc0 + kk) * 4096 + n0 + nn];
    }
    __syncthreads();
    // A: k-tap constant across the whole 32-K chunk (chunks stay inside one tap)
    const int k   = kc0 >> 7;
    const int c0  = (kc0 & 127) + 16 * half;
    const bf16_t* ap = &xs[(row + k) * XS_LD + c0];
    const bf16_t* bp = &bsh[(wave * 16 + (lane & 15)) * BS_LD + 16 * half];
    acc = wmma_bf16(frag2(ap, ap + 8), frag2(bp, bp + 8), acc);
  }

  const int n = n0 + wave * 16 + (lane & 15);
  const float bias = b1[n];
#pragma unroll
  for (int v = 0; v < 8; ++v) {
    int t = t0 + v + 8 * half;
    float z = gelu_exact(acc[v] + bias);
    h1[((size_t)b * 2000 + t) * 4096 + n] = (bf16_t)z;
  }
}

// ---- Stage 2: conv2 (K=3, 4096->1024, stride 2, SAME) + GELU + transpose --
// GEMM: M=8000 (b*1000+t'), K=12288, N=1024. pad_left=0 (TF SAME, stride 2).
// Double-buffered LDS staging: one barrier per K-chunk.
__global__ __launch_bounds__(256) void conv2_gelu_tr_k(
    const bf16_t* __restrict__ h1,  // [8,2000,4096] bf16
    const float* __restrict__ w2,   // [12288,1024]
    const float* __restrict__ b2,   // [1024]
    bf16_t* __restrict__ h2t)       // [8,1024,1000] bf16 (transposed)
{
  __shared__ __align__(16) bf16_t as2[2][16 * AS_LD];
  __shared__ __align__(16) bf16_t bs2[2][128 * BS_LD];
  const int lane = threadIdx.x & 31;
  const int wave = threadIdx.x >> 5;
  const int m0 = blockIdx.y * 16;
  const int n0 = blockIdx.x * 128;
  const int row = lane & 15, half = lane >> 4;

  // stage one 32-K tile (A rows map each to their own (b,t'); k-tap per column)
  auto stage = [&](int buf, int kc0) {
    for (int idx = threadIdx.x; idx < 16 * 32; idx += 256) {
      int r = idx >> 5, kk = idx & 31;
      int m = m0 + r, bb_ = m / 1000, tp = m % 1000;
      int kc = kc0 + kk, k = kc >> 12, cin = kc & 4095;
      int tt = 2 * tp + k;  // input positions 2t', 2t'+1, 2t'+2
      as2[buf][r * AS_LD + kk] =
          (tt < 2000) ? h1[((size_t)bb_ * 2000 + tt) * 4096 + cin] : (bf16_t)0.f;
    }
    for (int idx = threadIdx.x; idx < 32 * 128; idx += 256) {
      int nn = idx & 127, kk = idx >> 7;
      bs2[buf][nn * BS_LD + kk] = (bf16_t)w2[(size_t)(kc0 + kk) * 1024 + n0 + nn];
    }
  };

  stage(0, 0);
  v8f acc = {};
  int cur = 0;
  for (int kc0 = 0; kc0 < 12288; kc0 += 32) {
    __syncthreads();
    if (kc0 + 32 < 12288) stage(cur ^ 1, kc0 + 32);
    if (kc0 + 128 < 12288)  // gfx1250 global_prefetch_b8, tile k+4 weights
      __builtin_prefetch(&w2[(size_t)(kc0 + 128) * 1024 + n0 + (threadIdx.x & 127)], 0, 1);
    const bf16_t* ap = &as2[cur][row * AS_LD + 16 * half];
    const bf16_t* bp = &bs2[cur][(wave * 16 + (lane & 15)) * BS_LD + 16 * half];
    acc = wmma_bf16(frag2(ap, ap + 8), frag2(bp, bp + 8), acc);
    cur ^= 1;
  }

  const int n = n0 + wave * 16 + (lane & 15);
  const float bias = b2[n];
#pragma unroll
  for (int v = 0; v < 8; ++v) {
    int m = m0 + v + 8 * half, bb_ = m / 1000, tp = m % 1000;
    float z = gelu_exact(acc[v] + bias);
    h2t[((size_t)bb_ * 1024 + n) * 1000 + tp] = (bf16_t)z;  // transposed store
  }
}

// ---- Stage 3: time-invariant LSTM input projection  Z0 = X @ Wk + b -------
// GEMM: M=8192 (b*1024+s), K=1000 (pad to 1024), N=2048. f32 output.
__global__ __launch_bounds__(256) void xwk_gemm_k(
    const bf16_t* __restrict__ h2t,  // [8192,1000] bf16
    const float* __restrict__ wk,    // [1000,2048]
    const float* __restrict__ lb,    // [2048]
    float* __restrict__ z0)          // [8192,2048]
{
  __shared__ __align__(16) bf16_t as[16 * AS_LD];
  __shared__ __align__(16) bf16_t bsh[128 * BS_LD];
  const int lane = threadIdx.x & 31;
  const int wave = threadIdx.x >> 5;
  const int m0 = blockIdx.y * 16;
  const int n0 = blockIdx.x * 128;
  const int row = lane & 15, half = lane >> 4;

  v8f acc = {};
  for (int k0 = 0; k0 < 1024; k0 += 32) {
    __syncthreads();
    for (int idx = threadIdx.x; idx < 16 * 32; idx += 256) {
      int r = idx >> 5, kk = idx & 31, k = k0 + kk;
      as[r * AS_LD + kk] =
          (k < 1000) ? h2t[(size_t)(m0 + r) * 1000 + k] : (bf16_t)0.f;
    }
    for (int idx = threadIdx.x; idx < 32 * 128; idx += 256) {
      int nn = idx & 127, kk = idx >> 7, k = k0 + kk;
      bsh[nn * BS_LD + kk] =
          (k < 1000) ? (bf16_t)wk[(size_t)k * 2048 + n0 + nn] : (bf16_t)0.f;
    }
    __syncthreads();
    const bf16_t* ap = &as[row * AS_LD + 16 * half];
    const bf16_t* bp = &bsh[(wave * 16 + (lane & 15)) * BS_LD + 16 * half];
    acc = wmma_bf16(frag2(ap, ap + 8), frag2(bp, bp + 8), acc);
  }

  const int n = n0 + wave * 16 + (lane & 15);
  const float bias = lb[n];
#pragma unroll
  for (int v = 0; v < 8; ++v) {
    int m = m0 + v + 8 * half;
    z0[(size_t)m * 2048 + n] = acc[v] + bias;
  }
}

// ---- Wr fp32 -> bf16, pre-swizzled into per-lane WMMA B-fragment order ----
// wrs[(((nt*16 + kc)*32 + lane)*16 + i)] = Wr[kc*32 + 16*(lane>>4) + i][nt*16 + (lane&15)]
// so each lane's whole fragment is one contiguous 32B global read.
__global__ void swizzle_wr_k(const float* __restrict__ wr,  // [512,2048]
                             bf16_t* __restrict__ wrs) {    // [1048576]
  int o = blockIdx.x * blockDim.x + threadIdx.x;
  if (o >= 128 * 16 * 32 * 16) return;
  int i  = o & 15;
  int l  = (o >> 4) & 31;
  int kc = (o >> 9) & 15;
  int nt = o >> 13;
  int k = kc * 32 + ((l >> 4) << 4) + i;
  int n = nt * 16 + (l & 15);
  wrs[o] = (bf16_t)wr[(size_t)k * 2048 + n];
}

// ---- Stage 4: sequential LSTM recurrence + gates + positional encoding ----
// Single workgroup (1024 threads = 32 waves). h (bf16) and c (f32) in LDS.
// Per step: zr = h @ Wr via WMMA (M-tile 16, rows 0..7 valid), gates, PE.
__global__ __launch_bounds__(1024) void lstm_seq_k(
    const float* __restrict__ z0,    // [8192,2048] = Z0 (x@Wk + b)
    const bf16_t* __restrict__ wrs,  // swizzled Wr fragments
    float* __restrict__ out)         // [8,1024,512]
{
  __shared__ __align__(16) bf16_t hb[16 * HB_LD];  // h, rows 8..15 stay zero
  __shared__ float cs[8 * 512];                    // c state
  __shared__ float zb[8 * 2048];                   // recurrent GEMM per step
  const int tid = threadIdx.x;
  const int lane = tid & 31, wave = tid >> 5;
  const int row = lane & 15, half = lane >> 4;

  for (int i = tid; i < 16 * HB_LD; i += 1024) hb[i] = (bf16_t)0.f;
  for (int i = tid; i < 8 * 512; i += 1024) cs[i] = 0.f;
  __syncthreads();

  for (int s = 0; s < 1024; ++s) {
    // --- recurrent GEMM: [16(8 valid) x 512] x [512 x 2048] ---
#pragma unroll
    for (int j = 0; j < 4; ++j) {
      const int nt = wave * 4 + j;              // 128 n-tiles over 32 waves
      const int n = nt * 16 + (lane & 15);
      v8f acc = {};
#pragma unroll
      for (int kc = 0; kc < 16; ++kc) {
        const bf16_t* ap = &hb[row * HB_LD + kc * 32 + 16 * half];
        const bf16_t* bp = &wrs[(size_t)(((nt * 16 + kc) * 32 + lane)) << 4];
        acc = wmma_bf16(frag2(ap, ap + 8), frag2(bp, bp + 8), acc);
      }
      if (half == 0) {                          // rows 0..7 (batch) only
#pragma unroll
        for (int v = 0; v < 8; ++v) zb[v * 2048 + n] = acc[v];
      }
    }
    __syncthreads();

    // --- gates (Keras order i,f,g,o), state update, output + PE ---
    for (int e = tid; e < 8 * 512; e += 1024) {
      const int r = e >> 9, hx = e & 511;
      const size_t zrow = ((size_t)r * 1024 + s) * 2048;
      float zi = zb[r * 2048 + hx]        + z0[zrow + hx];
      float zf = zb[r * 2048 + 512 + hx]  + z0[zrow + 512 + hx];
      float zg = zb[r * 2048 + 1024 + hx] + z0[zrow + 1024 + hx];
      float zo = zb[r * 2048 + 1536 + hx] + z0[zrow + 1536 + hx];
      float si = 1.f / (1.f + __expf(-zi));
      float sf = 1.f / (1.f + __expf(-zf));
      float so = 1.f / (1.f + __expf(-zo));
      float c = sf * cs[e] + si * tanhf(zg);
      cs[e] = c;
      float h = so * tanhf(c);
      hb[r * HB_LD + hx] = (bf16_t)h;
      // pe[s,hx] = sin/cos(s / 10000^(2*(hx/2)/512))
      float ex = (float)(2 * (hx >> 1)) * (1.0f / 512.0f);
      float ang = (float)s * __powf(10000.0f, -ex);
      float pe = (hx & 1) ? __cosf(ang) : __sinf(ang);
      out[((size_t)r * 1024 + s) * 512 + hx] = h + pe;
    }
    __syncthreads();
  }
}

// ---------------------------------------------------------------------------
extern "C" void kernel_launch(void* const* d_in, const int* in_sizes, int n_in,
                              void* d_out, int out_size, void* d_ws, size_t ws_size,
                              hipStream_t stream) {
  const float* x  = (const float*)d_in[0];  // [8,2000,128]
  const float* w1 = (const float*)d_in[1];  // [3,128,4096]
  const float* b1 = (const float*)d_in[2];  // [4096]
  const float* w2 = (const float*)d_in[3];  // [3,4096,1024]
  const float* b2 = (const float*)d_in[4];  // [1024]
  const float* wk = (const float*)d_in[5];  // [1000,2048]
  const float* wr = (const float*)d_in[6];  // [512,2048]
  const float* lb = (const float*)d_in[7];  // [2048]
  float* out = (float*)d_out;               // [8,1024,512]

  char* ws = (char*)d_ws;
  bf16_t* h1  = (bf16_t*)ws; ws += (size_t)8 * 2000 * 4096 * sizeof(bf16_t);
  bf16_t* h2t = (bf16_t*)ws; ws += (size_t)8 * 1024 * 1000 * sizeof(bf16_t);
  float*  z0  = (float*)ws;  ws += (size_t)8192 * 2048 * sizeof(float);
  bf16_t* wrs = (bf16_t*)ws; ws += (size_t)512 * 2048 * sizeof(bf16_t);

  conv1_gelu_k   <<<dim3(32, 1000), 256, 0, stream>>>(x, w1, b1, h1);
  conv2_gelu_tr_k<<<dim3(8, 500),   256, 0, stream>>>(h1, w2, b2, h2t);
  xwk_gemm_k     <<<dim3(16, 512),  256, 0, stream>>>(h2t, wk, lb, z0);
  swizzle_wr_k   <<<(512 * 2048 + 255) / 256, 256, 0, stream>>>(wr, wrs);
  lstm_seq_k     <<<1, 1024, 0, stream>>>(z0, wrs, out);
}
